// GRU_37546604101723
// MI455X (gfx1250) — matrix operand order repaired and checked by reading, not compiled, not run on hardware
//
#include <hip/hip_runtime.h>
#include <math.h>

typedef __attribute__((ext_vector_type(16))) __bf16 v16bf;
typedef __attribute__((ext_vector_type(8)))  __bf16 v8bf;
typedef __attribute__((ext_vector_type(8)))  float  v8f;

#define BSZ 2
#define S   4096
#define H   128
#define D   256
#define CF  384      // feature channels (H + D), WMMA K-dim
#define CIN 387      // 3 xyz + CF
#define KNN 4
#define O   128
#define ROWS (S*KNN) // 16384 rows per batch
#define EPS 1e-5f

// ---------------------------------------------------------------- xyz transpose
__global__ void xyzT_prep(const float* __restrict__ pc, float* __restrict__ xyzT) {
    int t = blockIdx.x * 256 + threadIdx.x;
    if (t >= BSZ * S * 3) return;
    int d = t % 3; int s = (t / 3) % S; int b = t / (3 * S);
    xyzT[t] = pc[((size_t)b * 3 + d) * S + s];
}

// ---------------------------------------------------------------- brute-force KNN (k=4, includes self)
__global__ void knn_kernel(const float* __restrict__ xyzT, int* __restrict__ idx) {
    __shared__ float sx[512], sy[512], sz[512];
    int b = blockIdx.y;
    int q = blockIdx.x * 256 + threadIdx.x;
    const float* base = xyzT + (size_t)b * S * 3;
    float qx = base[q*3+0], qy = base[q*3+1], qz = base[q*3+2];
    float sqq = qx*qx + qy*qy + qz*qz;
    float bd[4] = {__builtin_inff(), __builtin_inff(), __builtin_inff(), __builtin_inff()};
    int   bi[4] = {0,0,0,0};
    for (int j0 = 0; j0 < S; j0 += 512) {
        __syncthreads();
        for (int e = threadIdx.x; e < 512*3; e += 256) {
            float v = base[j0*3 + e];
            int p = e / 3, dd = e % 3;
            if (dd == 0) sx[p] = v; else if (dd == 1) sy[p] = v; else sz[p] = v;
        }
        __syncthreads();
        for (int j = 0; j < 512; ++j) {
            float cx = sx[j], cy = sy[j], cz = sz[j];
            float sqc = cx*cx + cy*cy + cz*cz;
            float dot = qx*cx + qy*cy + qz*cz;
            float dst = sqq + sqc - 2.0f * dot;     // matches reference formula (self -> exactly 0)
            if (dst < bd[3]) {
                int pos = 3;
                while (pos > 0 && dst < bd[pos-1]) { bd[pos] = bd[pos-1]; bi[pos] = bi[pos-1]; --pos; }
                bd[pos] = dst; bi[pos] = j0 + j;    // strict < keeps earlier index on ties (stable like top_k)
            }
        }
    }
    for (int j = 0; j < 4; ++j) idx[((size_t)b * S + q) * 4 + j] = bi[j];
}

// ---------------------------------------------------------------- feature tables [B,S,384] bf16
__global__ void hxT_prep(const float* __restrict__ h, const float* __restrict__ x,
                         __bf16* __restrict__ out) {
    int t = blockIdx.x * 256 + threadIdx.x;
    if (t >= BSZ * S * CF) return;
    int c = t % CF; int s = (t / CF) % S; int b = t / (CF * S);
    float v = (c < H) ? h[((size_t)b * H + c) * S + s]
                      : x[((size_t)b * D + (c - H)) * S + s];
    out[t] = (__bf16)v;
}

__global__ void rhxT_prep(const float* __restrict__ h, const float* __restrict__ x,
                          const float* __restrict__ rbuf /*[B,S,H]*/, __bf16* __restrict__ out) {
    int t = blockIdx.x * 256 + threadIdx.x;
    if (t >= BSZ * S * CF) return;
    int c = t % CF; int s = (t / CF) % S; int b = t / (CF * S);
    float v;
    if (c < H) v = rbuf[((size_t)b * S + s) * H + c] * h[((size_t)b * H + c) * S + s];
    else       v = x[((size_t)b * D + (c - H)) * S + s];
    out[t] = (__bf16)v;
}

// ---------------------------------------------------------------- weight fragments
// frag[kch][nt][lane][e] = W[o = nt*16 + (lane&15)][3 + K],
// K = kch*32 + (lane<16 ? 0 : 8) + (e<8 ? e : e+8)   (B-matrix 32x16 bf16 per-lane layout)
__global__ void wfrag_prep(const float* __restrict__ W, __bf16* __restrict__ frag) {
    int t = blockIdx.x * 256 + threadIdx.x;
    if (t >= 12 * 8 * 32 * 16) return;
    int e    = t & 15;
    int lane = (t >> 4) & 31;
    int nt   = (t >> 9) & 7;
    int kch  = t >> 12;
    int o = nt * 16 + (lane & 15);
    int K = kch * 32 + ((lane & 16) ? 8 : 0) + ((e & 8) ? (e + 8) : e);
    frag[t] = (__bf16)W[(size_t)o * CIN + 3 + K];
}

// ---------------------------------------------------------------- gathered GEMM via WMMA bf16
// Block: 128 rows x 64 cols (col half chosen by blockIdx.z).
// Wave: 16-row strip x 4 column tiles. Manual double-buffer over K chunks.
__global__ __launch_bounds__(256) void gemm_gate(
    const __bf16* __restrict__ featT, const int* __restrict__ idx,
    const float* __restrict__ xyzT, const __bf16* __restrict__ wfrag,
    const float* __restrict__ W, const float* __restrict__ bias,
    float* __restrict__ y) {
    __shared__ int   snbr[128];
    __shared__ float srel[128][3];
    int b    = blockIdx.y;
    int row0 = blockIdx.x * 128;
    int ch   = blockIdx.z;          // column half: tiles ch*4 .. ch*4+3
    int tid  = threadIdx.x;
    if (tid < 128) {
        int r = row0 + tid;
        int s = r >> 2, kk = r & 3;
        int nb = idx[((size_t)b * S + s) * 4 + kk];
        snbr[tid] = nb;
        const float* xb = xyzT + (size_t)b * S * 3;
        srel[tid][0] = xb[nb*3+0] - xb[s*3+0];
        srel[tid][1] = xb[nb*3+1] - xb[s*3+1];
        srel[tid][2] = xb[nb*3+2] - xb[s*3+2];
    }
    __syncthreads();
    int wave = tid >> 5, lane = tid & 31;
    int arow = wave * 16 + (lane & 15);                 // A-matrix M = lane&15
    const __bf16* rowptr = featT + ((size_t)b * S + snbr[arow]) * CF;
    int off1 = (lane < 16) ? 0 : 8;                     // 16-bit 16x32 A layout K offsets
    const __bf16* wbase = wfrag + ((size_t)ch * 4 * 32 + lane) * 16;  // + kch*4096 + ntl*512

    v8f acc[4];
    #pragma unroll
    for (int t = 0; t < 4; ++t)
        #pragma unroll
        for (int j = 0; j < 8; ++j) acc[t][j] = 0.0f;

    // ---- prologue loads (chunk 0)
    v16bf A, Bf[4];
    {
        v8bf alo = *(const v8bf*)(rowptr + off1);
        v8bf ahi = *(const v8bf*)(rowptr + off1 + 16);
        #pragma unroll
        for (int i = 0; i < 8; ++i) { A[i] = alo[i]; A[8+i] = ahi[i]; }
        #pragma unroll
        for (int nt = 0; nt < 4; ++nt)
            Bf[nt] = *(const v16bf*)(wbase + nt * 32 * 16);
    }
    // ---- pipelined K loop (fully unrolled: 12 chunks)
    #pragma unroll
    for (int kc = 0; kc < CF; kc += 32) {
        v16bf An, Bn[4];
        if (kc + 32 < CF) {
            v8bf alo = *(const v8bf*)(rowptr + kc + 32 + off1);
            v8bf ahi = *(const v8bf*)(rowptr + kc + 32 + off1 + 16);
            #pragma unroll
            for (int i = 0; i < 8; ++i) { An[i] = alo[i]; An[8+i] = ahi[i]; }
            const __bf16* wp = wbase + (size_t)((kc >> 5) + 1) * 8 * 32 * 16;
            #pragma unroll
            for (int nt = 0; nt < 4; ++nt)
                Bn[nt] = *(const v16bf*)(wp + nt * 32 * 16);
        }
        #pragma unroll
        for (int nt = 0; nt < 4; ++nt)
            acc[nt] = __builtin_amdgcn_wmma_f32_16x16x32_bf16(
                false, A, false, Bf[nt], (short)0, acc[nt], false, false);
        A = An;
        #pragma unroll
        for (int nt = 0; nt < 4; ++nt) Bf[nt] = Bn[nt];
    }
    // ---- epilogue: xyz channels + bias, store fp32
    #pragma unroll
    for (int nt = 0; nt < 4; ++nt) {
        int col = (ch * 4 + nt) * 16 + (lane & 15);
        float w0 = W[(size_t)col * CIN + 0];
        float w1 = W[(size_t)col * CIN + 1];
        float w2 = W[(size_t)col * CIN + 2];
        float bb = bias[col];
        #pragma unroll
        for (int j = 0; j < 8; ++j) {
            int orow = wave * 16 + j + ((lane < 16) ? 0 : 8);   // C/D layout M
            float val = acc[nt][j] + w0 * srel[orow][0] + w1 * srel[orow][1]
                                   + w2 * srel[orow][2] + bb;
            y[(((size_t)b * ROWS) + row0 + orow) * O + col] = val;
        }
    }
}

// ---------------------------------------------------------------- instance-norm stats
__global__ void zero_stats(float* __restrict__ sum, float* __restrict__ sumsq) {
    int t = threadIdx.x;
    if (t < BSZ * O) { sum[t] = 0.0f; sumsq[t] = 0.0f; }
}

__global__ void stats_partial(const float* __restrict__ y,
                              float* __restrict__ sum, float* __restrict__ sumsq) {
    __shared__ float ls[256], ls2[256];
    int bid = blockIdx.x;            // B*128 blocks
    int b = bid >> 7, rb = bid & 127;
    int t = threadIdx.x;
    int col = t & 127, half = t >> 7;
    float s = 0.0f, s2 = 0.0f;
    const float* yb = y + (((size_t)b * ROWS) + (size_t)rb * 128) * O;
    for (int row = half; row < 128; row += 2) {
        float v = yb[(size_t)row * O + col];
        s += v; s2 += v * v;
    }
    ls[t] = s; ls2[t] = s2;
    __syncthreads();
    if (t < 128) {
        atomicAdd(&sum[b * O + col],   ls[t] + ls[t + 128]);
        atomicAdd(&sumsq[b * O + col], ls2[t] + ls2[t + 128]);
    }
}

__global__ void stats_finalize(const float* __restrict__ sum, const float* __restrict__ sumsq,
                               float* __restrict__ mean, float* __restrict__ invstd) {
    int t = threadIdx.x;
    if (t >= BSZ * O) return;
    float m = sum[t] * (1.0f / (float)ROWS);
    float v = sumsq[t] * (1.0f / (float)ROWS) - m * m;
    mean[t] = m;
    invstd[t] = rsqrtf(v + EPS);
}

// ---------------------------------------------------------------- normalize + max-pool + sigmoid -> gate [B,S,O]
__global__ void finalize_sig(const float* __restrict__ y, const float* __restrict__ mean,
                             const float* __restrict__ invstd, float* __restrict__ dst) {
    int t = blockIdx.x * 256 + threadIdx.x;
    if (t >= BSZ * S * O) return;
    int o = t & 127; int s = (t >> 7) & 4095; int b = t >> 19;
    float m = mean[b * O + o], is = invstd[b * O + o];
    const float* yp = y + (((size_t)b * ROWS) + (size_t)s * 4) * O + o;
    float v0 = (yp[0]     - m) * is;
    float v1 = (yp[O]     - m) * is;
    float v2 = (yp[2*O]   - m) * is;
    float v3 = (yp[3*O]   - m) * is;
    float mx = fmaxf(fmaxf(v0, v1), fmaxf(v2, v3));
    dst[t] = 1.0f / (1.0f + expf(-mx));
}

// ---------------------------------------------------------------- q gate + GRU combine
__global__ void finalize_q(const float* __restrict__ y, const float* __restrict__ mean,
                           const float* __restrict__ invstd, const float* __restrict__ zbuf,
                           const float* __restrict__ h, float* __restrict__ out) {
    int t = blockIdx.x * 256 + threadIdx.x;
    if (t >= BSZ * S * O) return;
    int o = t & 127; int s = (t >> 7) & 4095; int b = t >> 19;
    float m = mean[b * O + o], is = invstd[b * O + o];
    const float* yp = y + (((size_t)b * ROWS) + (size_t)s * 4) * O + o;
    float v0 = (yp[0]   - m) * is;
    float v1 = (yp[O]   - m) * is;
    float v2 = (yp[2*O] - m) * is;
    float v3 = (yp[3*O] - m) * is;
    float q = tanhf(fmaxf(fmaxf(v0, v1), fmaxf(v2, v3)));
    float z = zbuf[t];
    float hv = h[((size_t)b * H + o) * S + s];
    out[((size_t)b * H + o) * S + s] = (1.0f - z) * hv + z * q;
}

// ---------------------------------------------------------------- launch
extern "C" void kernel_launch(void* const* d_in, const int* in_sizes, int n_in,
                              void* d_out, int out_size, void* d_ws, size_t ws_size,
                              hipStream_t stream) {
    const float* h  = (const float*)d_in[0];
    const float* x  = (const float*)d_in[1];
    const float* pc = (const float*)d_in[2];
    const float* Wz = (const float*)d_in[3];
    const float* bz = (const float*)d_in[4];
    const float* Wr = (const float*)d_in[5];
    const float* br = (const float*)d_in[6];
    const float* Wq = (const float*)d_in[7];
    const float* bq = (const float*)d_in[8];
    float* out = (float*)d_out;

    char* ws = (char*)d_ws;
    size_t off = 0;
    auto take = [&](size_t bytes) { size_t o = off; off = (off + bytes + 255) & ~(size_t)255; return o; };

    int*    idx    = (int*)   (ws + take((size_t)BSZ * S * 4 * sizeof(int)));
    float*  xyzT   = (float*) (ws + take((size_t)BSZ * S * 3 * sizeof(float)));
    __bf16* hxT    = (__bf16*)(ws + take((size_t)BSZ * S * CF * 2));
    __bf16* rhxT   = (__bf16*)(ws + take((size_t)BSZ * S * CF * 2));
    __bf16* fragZ  = (__bf16*)(ws + take((size_t)12 * 8 * 32 * 16 * 2));
    __bf16* fragR  = (__bf16*)(ws + take((size_t)12 * 8 * 32 * 16 * 2));
    __bf16* fragQ  = (__bf16*)(ws + take((size_t)12 * 8 * 32 * 16 * 2));
    float*  ybuf   = (float*) (ws + take((size_t)BSZ * ROWS * O * sizeof(float)));
    float*  ssum   = (float*) (ws + take((size_t)BSZ * O * sizeof(float)));
    float*  ssq    = (float*) (ws + take((size_t)BSZ * O * sizeof(float)));
    float*  smean  = (float*) (ws + take((size_t)BSZ * O * sizeof(float)));
    float*  sistd  = (float*) (ws + take((size_t)BSZ * O * sizeof(float)));
    float*  zbuf   = (float*) (ws + take((size_t)BSZ * S * O * sizeof(float)));
    float*  rbuf   = (float*) (ws + take((size_t)BSZ * S * O * sizeof(float)));

    // 1) geometry + KNN
    xyzT_prep<<<(BSZ * S * 3 + 255) / 256, 256, 0, stream>>>(pc, xyzT);
    knn_kernel<<<dim3(S / 256, BSZ), 256, 0, stream>>>(xyzT, idx);

    // 2) feature table + weight fragments
    hxT_prep<<<(BSZ * S * CF + 255) / 256, 256, 0, stream>>>(h, x, hxT);
    wfrag_prep<<<(12 * 8 * 32 * 16 + 255) / 256, 256, 0, stream>>>(Wz, fragZ);
    wfrag_prep<<<(12 * 8 * 32 * 16 + 255) / 256, 256, 0, stream>>>(Wr, fragR);
    wfrag_prep<<<(12 * 8 * 32 * 16 + 255) / 256, 256, 0, stream>>>(Wq, fragQ);

    dim3 ggrid(ROWS / 128, BSZ, 2);
    int nfin = (BSZ * S * O + 255) / 256;

    // 3) z gate
    gemm_gate<<<ggrid, 256, 0, stream>>>(hxT, idx, xyzT, fragZ, Wz, bz, ybuf);
    zero_stats<<<1, 256, 0, stream>>>(ssum, ssq);
    stats_partial<<<BSZ * 128, 256, 0, stream>>>(ybuf, ssum, ssq);
    stats_finalize<<<1, 256, 0, stream>>>(ssum, ssq, smean, sistd);
    finalize_sig<<<nfin, 256, 0, stream>>>(ybuf, smean, sistd, zbuf);

    // 4) r gate
    gemm_gate<<<ggrid, 256, 0, stream>>>(hxT, idx, xyzT, fragR, Wr, br, ybuf);
    zero_stats<<<1, 256, 0, stream>>>(ssum, ssq);
    stats_partial<<<BSZ * 128, 256, 0, stream>>>(ybuf, ssum, ssq);
    stats_finalize<<<1, 256, 0, stream>>>(ssum, ssq, smean, sistd);
    finalize_sig<<<nfin, 256, 0, stream>>>(ybuf, smean, sistd, rbuf);

    // 5) q gate from [r*h, x]
    rhxT_prep<<<(BSZ * S * CF + 255) / 256, 256, 0, stream>>>(h, x, rbuf, rhxT);
    gemm_gate<<<ggrid, 256, 0, stream>>>(rhxT, idx, xyzT, fragQ, Wq, bq, ybuf);
    zero_stats<<<1, 256, 0, stream>>>(ssum, ssq);
    stats_partial<<<BSZ * 128, 256, 0, stream>>>(ybuf, ssum, ssq);
    stats_finalize<<<1, 256, 0, stream>>>(ssum, ssq, smean, sistd);
    finalize_q<<<nfin, 256, 0, stream>>>(ybuf, smean, sistd, zbuf, h, out);
}
